// SelfAttention_33449205301776
// MI455X (gfx1250) — matrix operand order repaired and checked by reading, not compiled, not run on hardware
//
#include <hip/hip_runtime.h>

typedef __attribute__((ext_vector_type(16))) _Float16 v16h;
typedef __attribute__((ext_vector_type(8)))  _Float16 v8h;
typedef __attribute__((ext_vector_type(8)))  float    v8f;
typedef __attribute__((ext_vector_type(4)))  float    f4;
typedef __attribute__((ext_vector_type(2)))  float    f2;

#define WMMA_F16(a, b, c) \
  __builtin_amdgcn_wmma_f32_16x16x32_f16(false, (a), false, (b), (short)0, (c), false, false)

// x: (8,8,1024,192) f32; channels [0,64)=Q, [64,128)=K, [128,192)=V
// out: (8,8,1024,64) f32
// Block = 4 waves = 4 consecutive 16-row q-tiles of ONE head; K/V chunks are
// staged once per block into double-buffered LDS (f16, V transposed) and
// shared by all 4 waves. Flash-attention over 32-wide KV chunks.
__global__ __launch_bounds__(128)
void attn1250(const float* __restrict__ x, float* __restrict__ out)
{
  constexpr int N = 1024, C = 192, D = 64;
  __shared__ _Float16 kl[2][32][64];   // K chunk, row-major f16   (4 KB x2)
  __shared__ _Float16 vt[2][64][32];   // V chunk, f16 transposed  (4 KB x2)

  const int t    = threadIdx.x;
  const int lane = t & 31;
  const int wv   = t >> 5;
  const int gid  = blockIdx.x * 4 + wv;       // 4096 waves total
  const int bp   = gid >> 6;                  // head 0..63 (same for all 4 waves)
  const int q0   = (gid & 63) << 4;           // q-tile start row
  const int hi   = lane >> 4;
  const int m16  = lane & 15;
  const float scale = 1.0f / 12.0f;           // 144^-0.5

  const float* xb = x + (size_t)bp * N * C;
  const v8f z8 = {0.f,0.f,0.f,0.f,0.f,0.f,0.f,0.f};

  // staging work assignment (128 threads)
  const int krow = t >> 2;                    // K: row 0..31
  const int kc0  = (t & 3) << 4;              // K: col 0,16,32,48
  const int vd0  = (t & 31) << 1;             // V: d pair 0,2,..,62
  const int vkv  = (t >> 5) << 3;             // V: kv base 0,8,16,24

  // ---- Q^T as B operands for S^T = K * Q^T (pre-scaled) ----
  // B(32x16): value(kk,n) -> lane = n + 16*(kk>=16), elem e = kk&15
  v16h bq[2];
  {
    const float* qr = xb + (size_t)(q0 + m16) * C;
    #pragma unroll
    for (int fs = 0; fs < 2; ++fs) {
      const float* p = qr + fs * 32 + hi * 16;
      #pragma unroll
      for (int i = 0; i < 4; ++i) {
        f4 v = *(const f4*)(p + 4 * i);
        bq[fs][4*i+0] = (_Float16)(v.x * scale);
        bq[fs][4*i+1] = (_Float16)(v.y * scale);
        bq[fs][4*i+2] = (_Float16)(v.z * scale);
        bq[fs][4*i+3] = (_Float16)(v.w * scale);
      }
    }
  }

  v8f o[4] = { z8, z8, z8, z8 };              // O accumulator: 16q x 64d
  float m_run = -3.0e38f, l_run = 0.0f;

  int buf = 0;
  for (int kv0 = 0; kv0 < N; kv0 += 32, buf ^= 1) {
    // ======== stage K chunk (coalesced float4, cvt once, b128 LDS stores) ====
    {
      const float* kp = xb + (size_t)(kv0 + krow) * C + 64 + kc0;
      f4 a = *(const f4*)(kp);
      f4 b = *(const f4*)(kp + 4);
      f4 c = *(const f4*)(kp + 8);
      f4 d = *(const f4*)(kp + 12);
      v8h h0 = { (_Float16)a.x, (_Float16)a.y, (_Float16)a.z, (_Float16)a.w,
                 (_Float16)b.x, (_Float16)b.y, (_Float16)b.z, (_Float16)b.w };
      v8h h1 = { (_Float16)c.x, (_Float16)c.y, (_Float16)c.z, (_Float16)c.w,
                 (_Float16)d.x, (_Float16)d.y, (_Float16)d.z, (_Float16)d.w };
      *(v8h*)&kl[buf][krow][kc0]     = h0;
      *(v8h*)&kl[buf][krow][kc0 + 8] = h1;
    }
    // ======== stage V chunk transposed (coalesced float2 rows -> d-major) ====
    {
      const float* vp = xb + (size_t)(kv0 + vkv) * C + 128 + vd0;
      v8h h0, h1;
      #pragma unroll
      for (int i = 0; i < 8; ++i) {
        f2 v = *(const f2*)(vp + (size_t)i * C);
        h0[i] = (_Float16)v.x;
        h1[i] = (_Float16)v.y;
      }
      *(v8h*)&vt[buf][vd0]    [vkv] = h0;
      *(v8h*)&vt[buf][vd0 + 1][vkv] = h1;
    }
    __syncthreads();   // double buffering -> single barrier per chunk

    // ======== S^T chunk (32kv x 16q) as two C blocks ========
    v8f st[2] = { z8, z8 };
    #pragma unroll
    for (int b = 0; b < 2; ++b) {
      // A(16x32): lane row m16; e0..15 -> f = fs*32 + hi*16 + e
      const _Float16* krw = &kl[buf][16*b + m16][hi * 16];
      #pragma unroll
      for (int fs = 0; fs < 2; ++fs) {
        v8h lo = *(const v8h*)(krw + fs * 32);
        v8h hv = *(const v8h*)(krw + fs * 32 + 8);
        v16h ak = __builtin_shufflevector(lo, hv,
                    0,1,2,3,4,5,6,7,8,9,10,11,12,13,14,15);
        st[b] = WMMA_F16(ak, bq[fs], st[b]);
      }
    }

    // ======== V B operands: two 16B reads from transposed LDS ========
    // B elem e<8: kk = hi*8 + e, e>=8: kk = 16 + hi*8 + (e-8); col d = db*16 + m16
    v16h bv[4];
    #pragma unroll
    for (int db = 0; db < 4; ++db)
      bv[db] = __builtin_shufflevector(*(const v8h*)&vt[buf][db * 16 + m16][hi * 8],
                                       *(const v8h*)&vt[buf][db * 16 + m16][16 + hi * 8],
                                       0,1,2,3,4,5,6,7,8,9,10,11,12,13,14,15);

    // ======== online softmax: lane holds S[q=m16][kv = 16b + 8hi + r] ========
    float smax = fmaxf(st[0][0], st[1][0]);
    #pragma unroll
    for (int r = 1; r < 8; ++r) smax = fmaxf(smax, fmaxf(st[0][r], st[1][r]));
    smax = fmaxf(smax, __shfl_xor(smax, 16, 32));   // combine lane pair (m, m+16)
    const float mnew = fmaxf(m_run, smax);
    const float corr = __expf(m_run - mnew);

    float p0[8], p1[8], rsum = 0.0f;
    #pragma unroll
    for (int r = 0; r < 8; ++r) {
      p0[r] = __expf(st[0][r] - mnew);
      p1[r] = __expf(st[1][r] - mnew);
      rsum += p0[r] + p1[r];
    }
    rsum += __shfl_xor(rsum, 16, 32);
    l_run = l_run * corr + rsum;
    m_run = mnew;

    // P in C-layout of S^T == A-layout of P (no cross-lane movement needed)
    v16h ap;
    #pragma unroll
    for (int r = 0; r < 8; ++r) { ap[r] = (_Float16)p0[r]; ap[r + 8] = (_Float16)p1[r]; }

    // rescale O rows (row m = r + 8*hi at vgpr r); corr[m] sits at lane m
    #pragma unroll
    for (int r = 0; r < 8; ++r) {
      const float c = __shfl(corr, r + 8 * hi, 32);
      o[0][r] *= c; o[1][r] *= c; o[2][r] *= c; o[3][r] *= c;
    }
    #pragma unroll
    for (int db = 0; db < 4; ++db)
      o[db] = WMMA_F16(ap, bv[db], o[db]);
  }

  // ======== normalize and store (f32 output) ========
  const float linv = 1.0f / l_run;
  float* ob = out + (size_t)bp * N * D;
  #pragma unroll
  for (int r = 0; r < 8; ++r) {
    const float s = __shfl(linv, r + 8 * hi, 32);
    float* orow = ob + (size_t)(q0 + r + 8 * hi) * D + m16;
    orow[0]  = o[0][r] * s;
    orow[16] = o[1][r] * s;
    orow[32] = o[2][r] * s;
    orow[48] = o[3][r] * s;
  }
}

extern "C" void kernel_launch(void* const* d_in, const int* in_sizes, int n_in,
                              void* d_out, int out_size, void* d_ws, size_t ws_size,
                              hipStream_t stream) {
  (void)in_sizes; (void)n_in; (void)d_ws; (void)ws_size; (void)out_size;
  const float* x = (const float*)d_in[0];
  float* out = (float*)d_out;
  // 64 heads * 64 q-tiles = 4096 waves; 4 waves (one head) per 128-thread block
  attn1250<<<dim3(1024), dim3(128), 0, stream>>>(x, out);
}
